// RecurrentNetwork_88493506167254
// MI455X (gfx1250) — matrix operand order repaired and checked by reading, not compiled, not run on hardware
//
#include <hip/hip_runtime.h>
#include <hip/hip_bf16.h>
#include <stdint.h>

typedef __attribute__((ext_vector_type(16))) __bf16 v16bf;
typedef __attribute__((ext_vector_type(8)))  __bf16 v8bf;
typedef __attribute__((ext_vector_type(8)))  float  v8f;

#define B_    1024
#define NIN   512
#define NN    4096
#define NODES 4608   // NIN + NN
#define E_    65536

// ---- workspace layout (bytes) ----
#define W_OFF    ((size_t)0)
#define W_BYTES  ((size_t)NN * NODES * 2)      // 37,748,736  dense bf16 W [NN][NODES] row-major
#define XA_OFF   (W_OFF + W_BYTES)
#define X_BYTES  ((size_t)B_ * NODES * 2)      // 9,437,184   batch-major activ [B][NODES]
#define XB_OFF   (XA_OFF + X_BYTES)
#define CNT_OFF  (XB_OFF + X_BYTES)
#define CNT_BYTES ((size_t)NN * 4)
#define TOTAL_WS (CNT_OFF + CNT_BYTES)         // ~56.6 MB

// -------- zero workspace (capture-safe, deterministic) --------
__global__ void rn_zero_ws(uint4* p, size_t n16) {
  size_t i = (size_t)blockIdx.x * blockDim.x + threadIdx.x;
  size_t stride = (size_t)gridDim.x * blockDim.x;
  uint4 z; z.x = z.y = z.z = z.w = 0u;
  for (; i < n16; i += stride) p[i] = z;
}

// -------- inputs (B,NIN) f32 -> X[b][node] bf16, both buffers (coalesced both sides) ----
__global__ void rn_convert_inputs(const float* __restrict__ in,
                                  __bf16* __restrict__ Xa, __bf16* __restrict__ Xb) {
  int idx = blockIdx.x * blockDim.x + threadIdx.x;   // B*NIN threads
  int b = idx >> 9;           // batch
  int c = idx & 511;          // input node
  __bf16 v = (__bf16)in[idx];
  Xa[(size_t)b * NODES + c] = v;
  Xb[(size_t)b * NODES + c] = v;
}

__device__ __forceinline__ unsigned short rn_f2bf(float f) {
  union { float f; uint32_t u; } x; x.f = f;
  uint32_t u = x.u;
  uint32_t r = u + 0x7FFFu + ((u >> 16) & 1u);   // round-to-nearest-even
  return (unsigned short)(r >> 16);
}

// -------- densify: W[dst][src] += w  (duplicates sum via pk bf16 atomic) --------
__global__ void rn_scatter_edges(const float* __restrict__ w,
                                 const int* __restrict__ src, const int* __restrict__ dst,
                                 __bf16* __restrict__ W, int* __restrict__ cnt) {
  int e = blockIdx.x * blockDim.x + threadIdx.x;
  if (e >= E_) return;
  int s = src[e], d = dst[e];
  atomicAdd(&cnt[d], 1);
  unsigned short bf = rn_f2bf(w[e]);
  uint64_t addr  = (uint64_t)(uintptr_t)W + ((uint64_t)d * NODES + (uint64_t)s) * 2u;
  uint32_t val   = (addr & 2u) ? ((uint32_t)bf << 16) : (uint32_t)bf;  // other half adds +0
  uint64_t waddr = addr & ~3ull;
  asm volatile("global_atomic_pk_add_bf16 %0, %1, off"
               :: "v"(waddr), "v"(val) : "memory");
}

// -------- one recurrent pass as dense bf16 GEMM + tanh epilogue --------
// C[M=4096][N=1024] = W[M][K] * X[N][K]^T ; Xn[n][NIN+m] = mask*tanh(bias+resp*C)
#define BM 128
#define BN 128
#define KB 32
#define LPAD 8   // row pad (bf16 elems) to spread LDS banks
#define NK (NODES / KB)   // 144

__global__ void __launch_bounds__(256)
rn_gemm_pass(const __bf16* __restrict__ W, const __bf16* __restrict__ X,
             __bf16* __restrict__ Xn,
             const float* __restrict__ bias, const float* __restrict__ resp,
             const int* __restrict__ cnt, float* __restrict__ out /* null except last pass */) {
  // double-buffered A tile: [buf][m][k]
  __shared__ __attribute__((aligned(16))) __bf16 At[2][BM][KB + LPAD];

  const int tid   = threadIdx.x;
  const int lane  = tid & 31;
  const int wave  = tid >> 5;
  const int waveM = wave & 3;   // 4 waves along M
  const int waveN = wave >> 2;  // 2 waves along N
  const int mBase = blockIdx.y * BM;
  const int nBase = blockIdx.x * BN;

  const int lrow  = lane & 15;
  const int khalf = lane >> 4;   // 0/1

  // per-thread async A-tile chunk coords (512 chunks of 16B per tile, 2 per thread)
  const int r0 = tid >> 2,          c0 = tid & 3;
  const int r1 = (tid + 256) >> 2,  c1 = (tid + 256) & 3;
  const uint32_t At0 = (uint32_t)(uintptr_t)&At[0][0][0];
  const uint32_t At1 = (uint32_t)(uintptr_t)&At[1][0][0];

  v8f acc[2][4];
  #pragma unroll
  for (int i = 0; i < 2; ++i)
    #pragma unroll
    for (int j = 0; j < 4; ++j)
      #pragma unroll
      for (int q = 0; q < 8; ++q) acc[i][j][q] = 0.0f;

  // issue async loads for tile 0
  {
    uint64_t g0 = (uint64_t)(uintptr_t)(W + (size_t)(mBase + r0) * NODES + c0 * 8);
    uint64_t g1 = (uint64_t)(uintptr_t)(W + (size_t)(mBase + r1) * NODES + c1 * 8);
    uint32_t l0 = At0 + (uint32_t)(r0 * (KB + LPAD) * 2 + c0 * 16);
    uint32_t l1 = At0 + (uint32_t)(r1 * (KB + LPAD) * 2 + c1 * 16);
    asm volatile("global_load_async_to_lds_b128 %0, %1, off" :: "v"(l0), "v"(g0) : "memory");
    asm volatile("global_load_async_to_lds_b128 %0, %1, off" :: "v"(l1), "v"(g1) : "memory");
  }

  for (int it = 0; it < NK; ++it) {
    const int k0 = it * KB;
    const int cur = it & 1;
    // prefetch next A tile into the other buffer (prev readers passed last barrier)
    if (it + 1 < NK) {
      const uint32_t Ab = (cur ? At0 : At1);
      uint64_t g0 = (uint64_t)(uintptr_t)(W + (size_t)(mBase + r0) * NODES + (k0 + KB) + c0 * 8);
      uint64_t g1 = (uint64_t)(uintptr_t)(W + (size_t)(mBase + r1) * NODES + (k0 + KB) + c1 * 8);
      uint32_t l0 = Ab + (uint32_t)(r0 * (KB + LPAD) * 2 + c0 * 16);
      uint32_t l1 = Ab + (uint32_t)(r1 * (KB + LPAD) * 2 + c1 * 16);
      asm volatile("global_load_async_to_lds_b128 %0, %1, off" :: "v"(l0), "v"(g0) : "memory");
      asm volatile("global_load_async_to_lds_b128 %0, %1, off" :: "v"(l1), "v"(g1) : "memory");
      asm volatile("s_wait_asynccnt 2" ::: "memory");  // tile `it` complete, next in flight
    } else {
      asm volatile("s_wait_asynccnt 0" ::: "memory");
    }

    // B fragments straight from global (batch-major X: each lane 32B contiguous, L2-hot)
    v16bf b[4];
    #pragma unroll
    for (int j = 0; j < 4; ++j) {
      const __bf16* bp = X + (size_t)(nBase + waveN * 64 + j * 16 + lrow) * NODES
                           + k0 + khalf * 16;
      v8bf lo = *(const v8bf*)bp;          // K = khalf*16 + 0..7
      v8bf hi = *(const v8bf*)(bp + 8);    // K = khalf*16 + 8..15
      #pragma unroll
      for (int q = 0; q < 8; ++q) { b[j][q] = lo[q]; b[j][q + 8] = hi[q]; }
    }

    __syncthreads();   // all waves' async tile-`it` loads visible

    v16bf a[2];
    #pragma unroll
    for (int i = 0; i < 2; ++i) {
      const __bf16* ap = &At[cur][waveM * 32 + i * 16 + lrow][khalf * 16];
      v8bf lo = *(const v8bf*)ap;          // K = khalf*16 + 0..7
      v8bf hi = *(const v8bf*)(ap + 8);    // K = khalf*16 + 8..15
      #pragma unroll
      for (int q = 0; q < 8; ++q) { a[i][q] = lo[q]; a[i][q + 8] = hi[q]; }
    }

    #pragma unroll
    for (int i = 0; i < 2; ++i)
      #pragma unroll
      for (int j = 0; j < 4; ++j)
        acc[i][j] = __builtin_amdgcn_wmma_f32_16x16x32_bf16(
            false, a[i], false, b[j], (short)0, acc[i][j], false, false);

    __syncthreads();   // readers done before buffer is async-overwritten next iter
  }

  // ---- epilogue: act = tanh(bias + resp*acc), mask, vectorized bf16 state store ----
  const int mhalf = (lane >> 4) * 8;
  #pragma unroll
  for (int i = 0; i < 2; ++i) {
    const int mb = mBase + waveM * 32 + i * 16 + mhalf;   // multiple of 8
    float bi[8], re[8]; int ct[8];
    #pragma unroll
    for (int r = 0; r < 8; ++r) { bi[r] = bias[mb + r]; re[r] = resp[mb + r]; ct[r] = cnt[mb + r]; }
    #pragma unroll
    for (int j = 0; j < 4; ++j) {
      const int n = nBase + waveN * 64 + j * 16 + lrow;
      v8bf sv; float ov[8];
      #pragma unroll
      for (int r = 0; r < 8; ++r) {
        float v = tanhf(bi[r] + re[r] * acc[i][j][r]);
        v = (ct[r] > 0) ? v : 0.0f;
        sv[r] = (__bf16)v; ov[r] = v;
      }
      *(v8bf*)(Xn + (size_t)n * NODES + NIN + mb) = sv;    // 16B store, next pass's B operand
      if (out != nullptr && mb < 256) {
        #pragma unroll
        for (int r = 0; r < 8; ++r) out[(size_t)n * 256 + mb + r] = ov[r];
      }
    }
  }
}

extern "C" void kernel_launch(void* const* d_in, const int* in_sizes, int n_in,
                              void* d_out, int out_size, void* d_ws, size_t ws_size,
                              hipStream_t stream) {
  (void)in_sizes; (void)n_in; (void)out_size; (void)ws_size;
  const float* inputs  = (const float*)d_in[0];
  const float* weights = (const float*)d_in[1];
  const float* bias    = (const float*)d_in[2];
  const float* resp    = (const float*)d_in[3];
  const int*   src     = (const int*)d_in[4];
  const int*   dst     = (const int*)d_in[5];
  // d_in[6]=n_passes(3), d_in[7]=n_outputs(256): fixed by setup, baked in.
  float* out = (float*)d_out;

  char* ws = (char*)d_ws;
  __bf16* W   = (__bf16*)(ws + W_OFF);
  __bf16* Xa  = (__bf16*)(ws + XA_OFF);
  __bf16* Xb  = (__bf16*)(ws + XB_OFF);
  int*    cnt = (int*)(ws + CNT_OFF);

  rn_zero_ws<<<2048, 256, 0, stream>>>((uint4*)ws, TOTAL_WS / 16);
  rn_convert_inputs<<<(B_ * NIN) / 256, 256, 0, stream>>>(inputs, Xa, Xb);
  rn_scatter_edges<<<E_ / 256, 256, 0, stream>>>(weights, src, dst, W, cnt);

  dim3 grid(B_ / BN, NN / BM);   // (8, 32)
  // pass 0: Xa -> Xb ; pass 1: Xb -> Xa ; pass 2: Xa -> Xb (+ final output)
  rn_gemm_pass<<<grid, 256, 0, stream>>>(W, Xa, Xb, bias, resp, cnt, nullptr);
  rn_gemm_pass<<<grid, 256, 0, stream>>>(W, Xb, Xa, bias, resp, cnt, nullptr);
  rn_gemm_pass<<<grid, 256, 0, stream>>>(W, Xa, Xb, bias, resp, cnt, out);
}
